// AttentiveTransformer_26955214749979
// MI455X (gfx1250) — compile-verified, matrix operand order checked by reading, not run on hardware
//
#include <hip/hip_runtime.h>
#include <hip/hip_bf16.h>

// Problem constants (from reference setup_inputs)
#define B_ROWS 16384
#define KDIM   512
#define FDIM   2048
#define VBS    128
#define BN_EPS 1e-5f

// Tiling
#define BM  128   // == VBS: one workgroup owns a full ghost-BN chunk
#define BN  128
#define KC  32    // bf16 WMMA K per step
#define TPB 256   // 8 waves of 32
#define NW  8

typedef __attribute__((ext_vector_type(16))) __bf16 v16bf;
typedef __attribute__((ext_vector_type(8)))  __bf16 v8bf;
typedef __attribute__((ext_vector_type(8)))  float  v8f;
typedef unsigned int u32x4 __attribute__((ext_vector_type(4)));
typedef int          i32x8 __attribute__((ext_vector_type(8)));
typedef int          i32x4 __attribute__((ext_vector_type(4)));

// ---------------- LDS arena layout (bytes) ----------------
// [     0, 32768)  fp32 A staging, double buffered: [2][128][32] f32
// [ 32768, 65536)  fp32 B staging, double buffered: [2][128][32] f32
// [ 65536, 73728)  bf16 A-hi  [128][32]
// [ 73728, 81920)  bf16 A-lo  [128][32]
// [ 81920, 90112)  bf16 B-hi  [128][32]
// [ 90112, 98304)  bf16 B-lo  [128][32]
// After the K loop everything above is dead; y-tile [128][129] f32
// (66048 B) overlays the arena from offset 0.
#define ARENA_BYTES 98304
#define OFF_FA 0
#define OFF_FB 32768
#define OFF_AH 65536
#define OFF_AL 73728
#define OFF_BH 81920
#define OFF_BL 90112

__device__ __forceinline__ unsigned pack2bf(__bf16 a, __bf16 b) {
    union { __bf16 h[2]; unsigned u; } t;
    t.h[0] = a; t.h[1] = b;
    return t.u;
}

// Build a v16bf fragment from a row of 32 bf16 in LDS, per ISA 7.12.2:
// lane-half 0 holds K = 0..7 and 16..23; lane-half 1 holds K = 8..15 and 24..31,
// each as an ascending, contiguous 8-element (16 B) run.
__device__ __forceinline__ v16bf frag16(const __bf16* rowp, int half) {
    v8bf r0 = *(const v8bf*)(rowp + 8 * half);
    v8bf r1 = *(const v8bf*)(rowp + 16 + 8 * half);
    return __builtin_shufflevector(r0, r1, 0, 1, 2, 3, 4, 5, 6, 7,
                                   8, 9, 10, 11, 12, 13, 14, 15);
}

// Issue one TDM 2-D tile load: KC x BM fp32 tile (row stride KDIM) -> LDS.
// D# per CDNA5 ISA ch.8: group0 {count|flags, lds_addr, gaddr_lo, gaddr_hi|type=2},
// group1 {data_size=4B, tensor_dim0=KDIM, tensor_dim1=rows, tile=32x128, stride0=KDIM}.
__device__ __forceinline__ void tdm_load_tile(unsigned lds_byte_off,
                                              const float* gptr,
                                              unsigned tensor_rows) {
    unsigned long long ga = (unsigned long long)(uintptr_t)gptr;
    u32x4 g0;
    g0[0] = 1u;                                     // count=1, user descriptor
    g0[1] = lds_byte_off;                           // LDS destination (bytes)
    g0[2] = (unsigned)ga;                           // global_addr[31:0]
    g0[3] = (unsigned)((ga >> 32) & 0x01FFFFFFull)  // global_addr[56:32]
            | 0x80000000u;                          // type=2 ("image")
    i32x8 g1;
    g1[0] = 0x00020000;                             // data_size=2 -> 4 bytes
    g1[1] = (int)((KDIM & 0xFFFF) << 16);           // tensor_dim0[15:0]
    g1[2] = (int)((KDIM >> 16) | ((tensor_rows & 0xFFFF) << 16));
    g1[3] = (int)((tensor_rows >> 16) | (KC << 16));// tile_dim0 = KC
    g1[4] = (int)BM;                                // tile_dim1 = 128, tile_dim2 = 0
    g1[5] = (int)KDIM;                              // tensor_dim0_stride[31:0]
    g1[6] = 0;
    g1[7] = 0;
    i32x4 z4 = {0, 0, 0, 0};
#if defined(__clang_major__) && (__clang_major__ >= 23)
    i32x8 z8 = {0, 0, 0, 0, 0, 0, 0, 0};
    __builtin_amdgcn_tensor_load_to_lds(g0, g1, z4, z4, z8, 0);
#else
    __builtin_amdgcn_tensor_load_to_lds(g0, g1, z4, z4, 0);
#endif
}

__global__ __launch_bounds__(TPB)
void gemm_gbn_kernel(const float* __restrict__ x,      // [B_ROWS, KDIM]
                     const float* __restrict__ W,      // [FDIM, KDIM]
                     const float* __restrict__ gamma,  // [FDIM]
                     const float* __restrict__ beta,   // [FDIM]
                     float* __restrict__ ynorm)        // [B_ROWS, FDIM]
{
    __shared__ __align__(16) char lds[ARENA_BYTES];
    __shared__ float sSum[2][BN];
    __shared__ float sSq[2][BN];
    __shared__ float sScale[BN];
    __shared__ float sShift[BN];

    float*  fA = (float*)(lds + OFF_FA);   // [2][128][32] f32
    float*  fB = (float*)(lds + OFF_FB);   // [2][128][32] f32
    __bf16* aH = (__bf16*)(lds + OFF_AH);  // [128][32]
    __bf16* aL = (__bf16*)(lds + OFF_AL);
    __bf16* bH = (__bf16*)(lds + OFF_BH);
    __bf16* bL = (__bf16*)(lds + OFF_BL);
    float*  yT = (float*)lds;              // [128][129] overlay after K loop

    const int tid  = threadIdx.x;
    const int w    = tid >> 5;
    const int lane = tid & 31;
    const int half = lane >> 4;
    const int lm   = lane & 15;
    const int m0   = blockIdx.y * BM;
    const int n0   = blockIdx.x * BN;

    // LDS byte offsets for the TDM descriptors (flat-addr low 32 bits == LDS offset).
    const unsigned arenaOff = (unsigned)(uintptr_t)lds;

    v8f acc[8];
#pragma unroll
    for (int i = 0; i < 8; i++) acc[i] = (v8f){0, 0, 0, 0, 0, 0, 0, 0};

    // Prologue: wave 0 kicks off the first K-slab via the Tensor Data Mover.
    if (w == 0) {
        tdm_load_tile(arenaOff + OFF_FA, x + (size_t)m0 * KDIM, (unsigned)B_ROWS);
        tdm_load_tile(arenaOff + OFF_FB, W + (size_t)n0 * KDIM, (unsigned)FDIM);
    }

    int cur = 0;
    for (int kc = 0; kc < KDIM; kc += KC) {
        if (w == 0) {
            if (kc + KC < KDIM) {
                // Issue next slab into the other buffer, then wait until only
                // those 2 remain outstanding (in-order => current slab landed).
                unsigned nb = cur ^ 1;
                tdm_load_tile(arenaOff + OFF_FA + nb * 16384,
                              x + (size_t)m0 * KDIM + kc + KC, (unsigned)B_ROWS);
                tdm_load_tile(arenaOff + OFF_FB + nb * 16384,
                              W + (size_t)n0 * KDIM + kc + KC, (unsigned)FDIM);
                __builtin_amdgcn_s_wait_tensorcnt(2);
            } else {
                __builtin_amdgcn_s_wait_tensorcnt(0);
            }
        }
        __syncthreads();   // current fp32 slab is in LDS for everyone

        // Split-convert fp32 -> bf16 hi/lo ONCE per tile (32 floats/thread).
#pragma unroll
        for (int j = 0; j < 4; j++) {
            int idx = tid + TPB * j;
            int row = idx >> 3;
            int kq  = (idx & 7) * 4;
            {
                float4 v = *(const float4*)&fA[(cur * BM + row) * KC + kq];
                __bf16 h0 = (__bf16)v.x, h1 = (__bf16)v.y,
                       h2 = (__bf16)v.z, h3 = (__bf16)v.w;
                *(uint2*)&aH[row * KC + kq] =
                    make_uint2(pack2bf(h0, h1), pack2bf(h2, h3));
                __bf16 l0 = (__bf16)(v.x - (float)h0), l1 = (__bf16)(v.y - (float)h1),
                       l2 = (__bf16)(v.z - (float)h2), l3 = (__bf16)(v.w - (float)h3);
                *(uint2*)&aL[row * KC + kq] =
                    make_uint2(pack2bf(l0, l1), pack2bf(l2, l3));
            }
            {
                float4 v = *(const float4*)&fB[(cur * BM + row) * KC + kq];
                __bf16 h0 = (__bf16)v.x, h1 = (__bf16)v.y,
                       h2 = (__bf16)v.z, h3 = (__bf16)v.w;
                *(uint2*)&bH[row * KC + kq] =
                    make_uint2(pack2bf(h0, h1), pack2bf(h2, h3));
                __bf16 l0 = (__bf16)(v.x - (float)h0), l1 = (__bf16)(v.y - (float)h1),
                       l2 = (__bf16)(v.z - (float)h2), l3 = (__bf16)(v.w - (float)h3);
                *(uint2*)&bL[row * KC + kq] =
                    make_uint2(pack2bf(l0, l1), pack2bf(l2, l3));
            }
        }
        __syncthreads();   // bf16 tiles ready; TDM fills the other buffer meanwhile

        // Fragments are now straight ds_load_b128 pairs — no VALU conversion here.
        const int arow = 16 * w + lm;
        v16bf a_hi = frag16(aH + arow * KC, half);
        v16bf a_lo = frag16(aL + arow * KC, half);

#pragma unroll
        for (int nt = 0; nt < 8; nt++) {
            const int bcol = 16 * nt + lm;
            v16bf b_hi = frag16(bH + bcol * KC, half);
            v16bf b_lo = frag16(bL + bcol * KC, half);
            // bf16x2 split product: hi*hi + hi*lo + lo*hi ~= fp32 GEMM
            acc[nt] = __builtin_amdgcn_wmma_f32_16x16x32_bf16(
                false, a_hi, false, b_hi, (short)0, acc[nt], false, false);
            acc[nt] = __builtin_amdgcn_wmma_f32_16x16x32_bf16(
                false, a_hi, false, b_lo, (short)0, acc[nt], false, false);
            acc[nt] = __builtin_amdgcn_wmma_f32_16x16x32_bf16(
                false, a_lo, false, b_hi, (short)0, acc[nt], false, false);
        }
        cur ^= 1;
    }

    __syncthreads();   // staging buffers dead; overlay y-tile on the arena
    // Spill accumulators per ISA C/D fragment mapping: VGPR r -> M=r+8*half.
#pragma unroll
    for (int nt = 0; nt < 8; nt++) {
#pragma unroll
        for (int r = 0; r < 8; r++) {
            int m = 16 * w + 8 * half + r;
            int n = 16 * nt + lm;
            yT[m * (BN + 1) + n] = acc[nt][r];
        }
    }
    __syncthreads();

    // Ghost-BN stats: mean/var per feature over the 128 rows (one virtual batch).
    {
        int col = tid & 127;
        int rh  = tid >> 7;
        float s = 0.f, q = 0.f;
#pragma unroll 8
        for (int r = 0; r < 64; r++) {
            float v = yT[(64 * rh + r) * (BN + 1) + col];
            s += v;
            q += v * v;
        }
        sSum[rh][col] = s;
        sSq[rh][col]  = q;
    }
    __syncthreads();
    if (tid < BN) {
        float s    = sSum[0][tid] + sSum[1][tid];
        float q    = sSq[0][tid] + sSq[1][tid];
        float mean = s * (1.0f / (float)VBS);
        float var  = q * (1.0f / (float)VBS) - mean * mean;
        float rstd = rsqrtf(var + BN_EPS);
        float sc   = gamma[n0 + tid] * rstd;
        sScale[tid] = sc;
        sShift[tid] = beta[n0 + tid] - mean * sc;
    }
    __syncthreads();

#pragma unroll 4
    for (int j = 0; j < 64; j++) {
        int e   = j * TPB + tid;
        int row = e >> 7;
        int col = e & 127;
        ynorm[(size_t)(m0 + row) * FDIM + n0 + col] =
            yT[row * (BN + 1) + col] * sScale[col] + sShift[col];
    }
}

// One block per row: z = ynorm*prior, sparsemax via Michelot's active-set iteration.
__global__ __launch_bounds__(TPB)
void sparsemax_kernel(const float* __restrict__ prior,  // [B_ROWS, FDIM]
                      float* __restrict__ y)            // in/out [B_ROWS, FDIM]
{
    __shared__ float sRedS[NW];
    __shared__ float sRedC[NW];
    __shared__ float sBcast[2];   // [0]=tau, [1]=k

    const int tid  = threadIdx.x;
    const int lane = tid & 31;
    const int w    = tid >> 5;
    const size_t row = blockIdx.x;

    float z[8];
#pragma unroll
    for (int i = 0; i < 8; i++) {
        size_t idx = row * FDIM + tid + TPB * i;
        z[i] = y[idx] * prior[idx];
    }

    // Initial tau = (sum(z) - 1) / D
    float s = 0.f;
#pragma unroll
    for (int i = 0; i < 8; i++) s += z[i];
#pragma unroll
    for (int o = 16; o > 0; o >>= 1) s += __shfl_xor(s, o, 32);
    if (lane == 0) sRedS[w] = s;
    __syncthreads();
    if (tid == 0) {
        float t = 0.f;
#pragma unroll
        for (int i = 0; i < NW; i++) t += sRedS[i];
        sBcast[0] = (t - 1.0f) / (float)FDIM;
        sBcast[1] = (float)FDIM;
    }
    __syncthreads();
    float tau   = sBcast[0];
    float kprev = sBcast[1];

    for (int it = 0; it < 32; ++it) {
        float S = 0.f, C = 0.f;
#pragma unroll
        for (int i = 0; i < 8; i++) {
            if (z[i] > tau) { S += z[i]; C += 1.0f; }
        }
#pragma unroll
        for (int o = 16; o > 0; o >>= 1) {
            S += __shfl_xor(S, o, 32);
            C += __shfl_xor(C, o, 32);
        }
        if (lane == 0) { sRedS[w] = S; sRedC[w] = C; }
        __syncthreads();
        if (tid == 0) {
            float St = 0.f, Ct = 0.f;
#pragma unroll
            for (int i = 0; i < NW; i++) { St += sRedS[i]; Ct += sRedC[i]; }
            Ct = fmaxf(Ct, 1.0f);
            sBcast[0] = (St - 1.0f) / Ct;
            sBcast[1] = Ct;
        }
        __syncthreads();
        tau = sBcast[0];
        float k = sBcast[1];
        if (k == kprev) break;   // uniform decision (LDS broadcast)
        kprev = k;
        __syncthreads();
    }

#pragma unroll
    for (int i = 0; i < 8; i++) {
        size_t idx = row * FDIM + tid + TPB * i;
        y[idx] = fmaxf(z[i] - tau, 0.0f);
    }
}

extern "C" void kernel_launch(void* const* d_in, const int* in_sizes, int n_in,
                              void* d_out, int out_size, void* d_ws, size_t ws_size,
                              hipStream_t stream) {
    (void)in_sizes; (void)n_in; (void)out_size; (void)d_ws; (void)ws_size;
    const float* x     = (const float*)d_in[0];   // [16384, 512]
    const float* prior = (const float*)d_in[1];   // [16384, 2048]
    const float* W     = (const float*)d_in[2];   // [2048, 512]
    const float* gamma = (const float*)d_in[3];   // [2048]
    const float* beta  = (const float*)d_in[4];   // [2048]
    float* out = (float*)d_out;                   // [16384, 2048]

    dim3 grid1(FDIM / BN, B_ROWS / BM);           // (16, 128)
    gemm_gbn_kernel<<<grid1, TPB, 0, stream>>>(x, W, gamma, beta, out);

    sparsemax_kernel<<<B_ROWS, TPB, 0, stream>>>(prior, out);
}
